// SSD_1949915153018
// MI455X (gfx1250) — matrix operand order repaired and checked by reading, not compiled, not run on hardware
//
#include <hip/hip_runtime.h>
#include <hip/hip_bf16.h>

typedef __attribute__((ext_vector_type(16))) _Float16 v16h;
typedef __attribute__((ext_vector_type(8)))  _Float16 v8h;
typedef __attribute__((ext_vector_type(8)))  float    v8f;
typedef __attribute__((ext_vector_type(4)))  unsigned int u32x4;
typedef __attribute__((ext_vector_type(8)))  int      i32x8;
typedef __attribute__((ext_vector_type(4)))  int      i32x4;

#define SSD_NUM_CLASSES 21
#define SSD_IOU_THR 0.5f
#define SSD_SCORE_THR 0.04f

// ---------------------------------------------------------------------------
// WMMA wrapper: D = A(16x32 f16) * B(32x16 f16) + C(16x16 f32)
// ---------------------------------------------------------------------------
__device__ inline v8f wmma16x16x32(v16h a, v16h b, v8f c) {
  return __builtin_amdgcn_wmma_f32_16x16x32_f16(
      /*neg_a=*/false, a, /*neg_b=*/false, b,
      /*c_mod=*/(short)0, c, /*reuse_a=*/false, /*reuse_b=*/false);
}

// ---------------------------------------------------------------------------
// Pack OIHW f32 weights into zero-padded row-major f16 A-matrix [CoutPad][K32]
// ---------------------------------------------------------------------------
__global__ void pack_weights(const float* __restrict__ w, _Float16* __restrict__ out,
                             int Cout, int K, int CoutPad, int K32) {
  int i = blockIdx.x * blockDim.x + threadIdx.x;
  int total = CoutPad * K32;
  if (i >= total) return;
  int co = i / K32, k = i % K32;
  float v = (co < Cout && k < K) ? w[(size_t)co * K + k] : 0.f;
  out[i] = (_Float16)v;
}

// ---------------------------------------------------------------------------
// Implicit-GEMM 3x3 conv via WMMA.
// Block = 256 threads = 8 waves; block tile = 64 Cout x 64 positions.
// Wave (coTile 0..3, pSeg 0..1) owns a 16(co) x 32(pos) D tile:
//   one A fragment, two B fragments, two v_wmma per K-step.
// LDS: As[64][40] f16 row-major [m][k] (row pitch 80B, 16B aligned chunks)
//      Bs[64][40] f16 row-major [pos][k] -> both fragments load as b128.
// Weight tile additionally staged by the Tensor Data Mover (D# with LDS
// padding matching the 40-half pitch); manual copy writes identical bytes.
// ---------------------------------------------------------------------------
template <typename InT, typename OutT, bool RELU>
__global__ __launch_bounds__(256) void conv3x3_wmma(
    const InT* __restrict__ in, const _Float16* __restrict__ wA,
    const float* __restrict__ bias, OutT* __restrict__ out,
    int Batch, int Cin, int H, int W,
    int Cout, int Ho, int Wo, int stride, int pad, int K, int K32)
{
  __shared__ __attribute__((aligned(16))) _Float16 As[64 * 40];
  __shared__ __attribute__((aligned(16))) _Float16 Bs[64 * 40];

  const int tid    = threadIdx.x;
  const int lane   = tid & 31;
  const int wv     = tid >> 5;
  const int coTile = wv & 3;      // 4 x 16 = 64 Cout
  const int pSeg   = wv >> 2;     // 2 x 32 = 64 positions
  const int lm     = lane & 15;
  const int half   = lane >> 4;

  const int total   = Batch * Ho * Wo;
  const int posBase = blockIdx.x * 64;
  const int coBase  = blockIdx.y * 64;
  const int CoutPad = gridDim.y * 64;

  // ---- fixed staging roles (hoisted out of the K loop) ----
  const int am = tid >> 2;            // A: row 0..63
  const int ak = (tid & 3) * 8;       // A: k chunk 0/8/16/24
  const int pp = tid >> 2;            // B: position 0..63
  const int kcol = (tid & 3) * 8;     // B: k chunk

  const int p = posBase + pp;
  const bool pvalid = (p < total);
  int ihbase = 0, iwbase = 0, bi = 0;
  if (pvalid) {
    int ow = p % Wo; int t2 = p / Wo; int oh = t2 % Ho; bi = t2 / Ho;
    ihbase = oh * stride - pad;
    iwbase = ow * stride - pad;
  }

  v8f acc0 = {}, acc1 = {};

  for (int kb = 0; kb < K32; kb += 32) {
#if defined(__gfx1250__)
    // ---- TDM: async-tensor load of the 64x32 f16 weight tile into As ----
    if (wv == 0) {
      unsigned long long ga =
          (unsigned long long)(const void*)(wA + (size_t)coBase * K32 + kb);
      unsigned int la = (unsigned int)(unsigned long long)(const void*)As;
      u32x4 g0;
      g0.x = 1u;                                   // count=1, user descriptor
      g0.y = la;                                   // lds_addr
      g0.z = (unsigned int)ga;                     // global_addr[31:0]
      g0.w = ((unsigned int)(ga >> 32) & 0x01FFFFFFu) | (2u << 30); // addr hi | type=2
      i32x8 g1;
      // wg_mask=0 | data_size=2B | pad_enable, interval=16dw, amount=4dw
      g1[0] = (int)((1u << 16) | (1u << 20) | (3u << 22) | (3u << 25));
      g1[1] = (int)(((unsigned)K32 & 0xFFFFu) << 16);   // tensor_dim0 lo16
      g1[2] = (int)((((unsigned)K32 >> 16) & 0xFFFFu) | ((unsigned)CoutPad << 16)); // dim0 hi | dim1 lo
      g1[3] = (int)(32u << 16);                    // dim1 hi=0 | tile_dim0=32
      g1[4] = 64;                                  // tile_dim1=64 | tile_dim2=0
      g1[5] = K32;                                 // tensor_dim0_stride lo32
      g1[6] = 0;                                   // stride0 hi | stride1 lo
      g1[7] = 0;
      i32x4 gz4 = {0, 0, 0, 0};
#if __clang_major__ >= 23
      i32x8 gz8 = {0, 0, 0, 0, 0, 0, 0, 0};
      __builtin_amdgcn_tensor_load_to_lds(g0, g1, gz4, gz4, gz8, 0);
#else
      __builtin_amdgcn_tensor_load_to_lds(g0, g1, gz4, gz4, 0);
#endif
    }
#endif

    // ---- stage A: 64x32 weight tile, one b128 per thread ----
    {
      v8h av = *(const v8h*)&wA[(size_t)(coBase + am) * K32 + kb + ak];
      *(v8h*)&As[am * 40 + ak] = av;
    }
    // ---- stage B (im2col, [pos][k]): 8 halves per thread, one b128 store ----
    {
      v8h bv;
#pragma unroll
      for (int j = 0; j < 8; ++j) {
        int kg = kb + kcol + j;
        _Float16 v = (_Float16)0.f;
        if (pvalid && kg < K) {
          int cin = kg / 9, rem = kg - cin * 9;
          int r = rem / 3, s = rem - r * 3;
          int ih = ihbase + r, iw = iwbase + s;
          if (ih >= 0 && ih < H && iw >= 0 && iw < W)
            v = (_Float16)(float)in[(size_t)((bi * Cin + cin) * H + ih) * W + iw];
        }
        bv[j] = v;
      }
      *(v8h*)&Bs[pp * 40 + kcol] = bv;
    }
    if (kb + 32 < K32)
      __builtin_prefetch(&wA[(size_t)coBase * K32 + (kb + 32)], 0, 0);
#if defined(__gfx1250__)
    if (wv == 0) __builtin_amdgcn_s_wait_tensorcnt(0);
#endif
    __syncthreads();

    // ---- fragments (contiguous k = half*16 + s, same in A and B) as b128 LDS loads ----
    union HV { v16h v; v8h h[2]; };
    HV af;
    const _Float16* arow = &As[(coTile * 16 + lm) * 40];
    af.h[0] = *(const v8h*)&arow[half * 16];       // k = half*16 .. +7
    af.h[1] = *(const v8h*)&arow[half * 16 + 8];   // k = half*16+8 .. +15

    HV bf0, bf1;
    const _Float16* brow0 = &Bs[(pSeg * 32 + lm) * 40];
    const _Float16* brow1 = &Bs[(pSeg * 32 + 16 + lm) * 40];
    bf0.h[0] = *(const v8h*)&brow0[half * 16];     // k = half*16 .. +7
    bf0.h[1] = *(const v8h*)&brow0[half * 16 + 8]; // k = half*16+8 .. +15
    bf1.h[0] = *(const v8h*)&brow1[half * 16];
    bf1.h[1] = *(const v8h*)&brow1[half * 16 + 8];

    acc0 = wmma16x16x32(af.v, bf0.v, acc0);
    acc1 = wmma16x16x32(af.v, bf1.v, acc1);
    __syncthreads();
  }

  // ---- epilogue: bias (+ReLU) and scatter the two D tiles ----
#pragma unroll
  for (int sub = 0; sub < 2; ++sub) {
    int p2 = posBase + pSeg * 32 + sub * 16 + lm;
    if (p2 < total) {
      int ow = p2 % Wo; int t2 = p2 / Wo; int oh = t2 % Ho; int bb = t2 / Ho;
#pragma unroll
      for (int r = 0; r < 8; ++r) {
        int co = coBase + coTile * 16 + half * 8 + r;   // C/D: M = half*8 + r
        if (co < Cout) {
          float v = (sub == 0 ? acc0[r] : acc1[r]) + bias[co];
          if (RELU) v = fmaxf(v, 0.f);
          out[(size_t)((bb * Cout + co) * Ho + oh) * Wo + ow] = (OutT)v;
        }
      }
    }
  }
}

// ---------------------------------------------------------------------------
// Per-anchor softmax (batch 7): score = max prob, label = argmax, gather box.
// ---------------------------------------------------------------------------
__global__ void head_post(const float* __restrict__ confs, const float* __restrict__ locs,
                          float* __restrict__ scores, int* __restrict__ labels,
                          float* __restrict__ boxes)
{
  int a = blockIdx.x * blockDim.x + threadIdx.x;
  if (a >= 1024) return;
  const int CB = 7 * SSD_NUM_CLASSES * 1024;
  float c[SSD_NUM_CLASSES];
  float m = -3.0e38f; int lab = 0;
#pragma unroll
  for (int j = 0; j < SSD_NUM_CLASSES; ++j) {
    c[j] = confs[CB + j * 1024 + a];
    if (c[j] > m) { m = c[j]; lab = j; }
  }
  float sum = 0.f;
#pragma unroll
  for (int j = 0; j < SSD_NUM_CLASSES; ++j) sum += expf(c[j] - m);
  scores[a] = 1.0f / sum;       // exp(max-max)/sum == max softmax prob
  labels[a] = lab;
  const int LB = 7 * 4 * 1024;
#pragma unroll
  for (int j = 0; j < 4; ++j) boxes[a * 4 + j] = locs[LB + j * 1024 + a];
}

// ---------------------------------------------------------------------------
// One block, 1024 threads: bitonic argsort (desc score, stable tie-break),
// sequential NMS with on-the-fly IoU, masked output writes.
// Output layout: boxes[1024*4] | labels[1024] | scores[1024] | confs[1024*21]
// ---------------------------------------------------------------------------
__global__ __launch_bounds__(1024) void sort_nms_write(
    const float* __restrict__ scores, const int* __restrict__ labels,
    const float* __restrict__ boxes, const float* __restrict__ confs,
    float* __restrict__ out)
{
  __shared__ float skey[1024];
  __shared__ int   sidx[1024];
  __shared__ float bx1[1024], by1[1024], bx2[1024], by2[1024];
  __shared__ float ssc[1024];
  __shared__ int   slab[1024];
  __shared__ int   keep[1024];
  __shared__ int   supp;

  const int tid = threadIdx.x;
  skey[tid] = scores[tid];
  sidx[tid] = tid;
  __syncthreads();

  for (int k = 2; k <= 1024; k <<= 1) {
    for (int j = k >> 1; j > 0; j >>= 1) {
      int p = tid ^ j;
      if (p > tid) {
        float ka = skey[tid], kb = skey[p];
        int   ia = sidx[tid], ib = sidx[p];
        bool desc    = ((tid & k) == 0);
        bool inOrder = (ka > kb) || (ka == kb && ia < ib);
        if (inOrder != desc) {
          skey[tid] = kb; skey[p] = ka;
          sidx[tid] = ib; sidx[p] = ia;
        }
      }
      __syncthreads();
    }
  }

  const int o = sidx[tid];
  const float x1 = boxes[o * 4 + 0], y1 = boxes[o * 4 + 1];
  const float x2 = boxes[o * 4 + 2], y2 = boxes[o * 4 + 3];
  bx1[tid] = x1; by1[tid] = y1; bx2[tid] = x2; by2[tid] = y2;
  ssc[tid]  = scores[o];
  slab[tid] = labels[o];
  keep[tid] = 0;
  __syncthreads();

  const float area  = (x2 - x1) * (y2 - y1);
  const int   valid = (ssc[tid] > SSD_SCORE_THR) ? 1 : 0;

  for (int i = 0; i < 1024; ++i) {
    if (tid == 0) supp = 0;
    __syncthreads();
    if (keep[tid]) {
      float ltx = fmaxf(bx1[i], x1), lty = fmaxf(by1[i], y1);
      float rbx = fminf(bx2[i], x2), rby = fminf(by2[i], y2);
      float inter = fmaxf(rbx - ltx, 0.f) * fmaxf(rby - lty, 0.f);
      float ai = (bx2[i] - bx1[i]) * (by2[i] - by1[i]);
      float iou = inter / (ai + area - inter + 1e-9f);
      if (iou > SSD_IOU_THR) atomicOr(&supp, 1);
    }
    __syncthreads();
    if (tid == i) keep[i] = (valid && supp == 0) ? 1 : 0;
    __syncthreads();
  }

  const int kq = keep[tid];
  float* obox = out;
  float* olab = out + 4096;
  float* osc  = out + 5120;
  float* ocf  = out + 6144;
  obox[tid * 4 + 0] = kq ? x1 : -1.f;
  obox[tid * 4 + 1] = kq ? y1 : -1.f;
  obox[tid * 4 + 2] = kq ? x2 : -1.f;
  obox[tid * 4 + 3] = kq ? y2 : -1.f;
  olab[tid] = kq ? (float)slab[tid] : 0.f;
  osc[tid]  = kq ? ssc[tid] : -1.f;
  const int CB = 7 * SSD_NUM_CLASSES * 1024;
#pragma unroll
  for (int j = 0; j < SSD_NUM_CLASSES; ++j)
    ocf[tid * SSD_NUM_CLASSES + j] = kq ? confs[CB + j * 1024 + o] : -1.f;
}

// ---------------------------------------------------------------------------
// Host orchestration
// ---------------------------------------------------------------------------
extern "C" void kernel_launch(void* const* d_in, const int* in_sizes, int n_in,
                              void* d_out, int out_size, void* d_ws, size_t ws_size,
                              hipStream_t stream)
{
  (void)in_sizes; (void)n_in; (void)out_size; (void)ws_size;

  const float* x  = (const float*)d_in[0];
  const float* w1 = (const float*)d_in[1];  const float* b1 = (const float*)d_in[2];
  const float* w2 = (const float*)d_in[3];  const float* b2 = (const float*)d_in[4];
  const float* w3 = (const float*)d_in[5];  const float* b3 = (const float*)d_in[6];
  const float* w4 = (const float*)d_in[7];  const float* b4 = (const float*)d_in[8];
  const float* wr = (const float*)d_in[9];  const float* br = (const float*)d_in[10];
  const float* wc = (const float*)d_in[11]; const float* bc = (const float*)d_in[12];

  char* ws = (char*)d_ws;
  size_t off = 0;
  auto alloc = [&](size_t bytes) -> char* {
    off = (off + 255) & ~(size_t)255;
    char* p = ws + off;
    off += bytes;
    return p;
  };
  auto cdiv = [](int a, int b) { return (a + b - 1) / b; };

  // padded f16 weight buffers  [CoutPad][K32]
  _Float16* wpk1 = (_Float16*)alloc((size_t)64  * 32   * 2);
  _Float16* wpk2 = (_Float16*)alloc((size_t)128 * 576  * 2);
  _Float16* wpk3 = (_Float16*)alloc((size_t)256 * 1152 * 2);
  _Float16* wpk4 = (_Float16*)alloc((size_t)256 * 2304 * 2);
  _Float16* wpkr = (_Float16*)alloc((size_t)64  * 2304 * 2);
  _Float16* wpkc = (_Float16*)alloc((size_t)64  * 2304 * 2);
  // f16 activations
  _Float16* h1 = (_Float16*)alloc((size_t)8 * 64  * 256 * 256 * 2);
  _Float16* h2 = (_Float16*)alloc((size_t)8 * 128 * 128 * 128 * 2);
  _Float16* h3 = (_Float16*)alloc((size_t)8 * 256 * 64  * 64  * 2);
  _Float16* h4 = (_Float16*)alloc((size_t)8 * 256 * 32  * 32  * 2);
  // f32 heads + post buffers
  float* locs   = (float*)alloc((size_t)8 * 4  * 1024 * 4);
  float* confs  = (float*)alloc((size_t)8 * 21 * 1024 * 4);
  float* scores = (float*)alloc(1024 * 4);
  int*   labels = (int*)  alloc(1024 * 4);
  float* boxesb = (float*)alloc(1024 * 4 * 4);

  // ---- weight packing ----
  pack_weights<<<cdiv(64  * 32,   256), 256, 0, stream>>>(w1, wpk1, 64,  27,   64,  32);
  pack_weights<<<cdiv(128 * 576,  256), 256, 0, stream>>>(w2, wpk2, 128, 576,  128, 576);
  pack_weights<<<cdiv(256 * 1152, 256), 256, 0, stream>>>(w3, wpk3, 256, 1152, 256, 1152);
  pack_weights<<<cdiv(256 * 2304, 256), 256, 0, stream>>>(w4, wpk4, 256, 2304, 256, 2304);
  pack_weights<<<cdiv(64  * 2304, 256), 256, 0, stream>>>(wr, wpkr, 4,   2304, 64,  2304);
  pack_weights<<<cdiv(64  * 2304, 256), 256, 0, stream>>>(wc, wpkc, 21,  2304, 64,  2304);

  // ---- backbone (stride-2, SAME with even H => pad_before = 0) ----
  conv3x3_wmma<float, _Float16, true><<<dim3(cdiv(8 * 256 * 256, 64), 1), 256, 0, stream>>>(
      x, wpk1, b1, h1, 8, 3, 512, 512, 64, 256, 256, 2, 0, 27, 32);
  conv3x3_wmma<_Float16, _Float16, true><<<dim3(cdiv(8 * 128 * 128, 64), 2), 256, 0, stream>>>(
      h1, wpk2, b2, h2, 8, 64, 256, 256, 128, 128, 128, 2, 0, 576, 576);
  conv3x3_wmma<_Float16, _Float16, true><<<dim3(cdiv(8 * 64 * 64, 64), 4), 256, 0, stream>>>(
      h2, wpk3, b3, h3, 8, 128, 128, 128, 256, 64, 64, 2, 0, 1152, 1152);
  conv3x3_wmma<_Float16, _Float16, true><<<dim3(cdiv(8 * 32 * 32, 64), 4), 256, 0, stream>>>(
      h3, wpk4, b4, h4, 8, 256, 64, 64, 256, 32, 32, 2, 0, 2304, 2304);

  // ---- heads (stride-1, SAME => pad 1), no ReLU, f32 out ----
  conv3x3_wmma<_Float16, float, false><<<dim3(cdiv(8 * 32 * 32, 64), 1), 256, 0, stream>>>(
      h4, wpkr, br, locs, 8, 256, 32, 32, 4, 32, 32, 1, 1, 2304, 2304);
  conv3x3_wmma<_Float16, float, false><<<dim3(cdiv(8 * 32 * 32, 64), 1), 256, 0, stream>>>(
      h4, wpkc, bc, confs, 8, 256, 32, 32, 21, 32, 32, 1, 1, 2304, 2304);

  // ---- post-processing (batch 7 only) ----
  head_post<<<4, 256, 0, stream>>>(confs, locs, scores, labels, boxesb);
  sort_nms_write<<<1, 1024, 0, stream>>>(scores, labels, boxesb, confs, (float*)d_out);
}
